// EdgeConvNet_4415226380301
// MI455X (gfx1250) — matrix operand order repaired and checked by reading, not compiled, not run on hardware
//
#include <hip/hip_runtime.h>

// ---------------------------------------------------------------------------
// EdgeConv x3 for MI455X (gfx1250): bf16 WMMA per-edge MLP + native
// global_atomic_max_num_f32 segment aggregation. x (25.6MB) is L2-resident
// (192MB L2), so the random gathers never touch HBM; the 236 GFLOP of MLP
// math runs on v_wmma_f32_16x16x32_bf16 with fp32 accumulation.
// ---------------------------------------------------------------------------

typedef __attribute__((ext_vector_type(16))) __bf16 bf16x16;
typedef __attribute__((ext_vector_type(8)))  __bf16 bf16x8;
typedef __attribute__((ext_vector_type(4)))  __bf16 bf16x4;
typedef __attribute__((ext_vector_type(8)))  float  f32x8;

#define CH        128                 // channels everywhere (C_IN = HID = C_OUT)
#define KDIM      256                 // 2*CH concat input of first Lin
#define A_STR     264                 // 256 + 8 bf16 pad (bank-conflict break)
#define H_STR     136                 // 128 + 8 bf16 pad
#define W1_STR    264
#define W2_STR    136
#define WAVES_PB  8                   // 256 threads / 32
#define NEG_HUGE  (-3.402823466e38f)  // -FLT_MAX sentinel for segment_max

// LDS budget (bf16 elems): W1t + W2t + 8*Atile + 8*Htile  (~200KB < 320KB/WGP)
#define LDS_ELEMS (128*W1_STR + 128*W2_STR + WAVES_PB*16*A_STR + WAVES_PB*16*H_STR)
#define LDS_BYTES (LDS_ELEMS * 2)

__device__ __forceinline__ void lds_fence() {
  // wave-local RAW fence for LDS (DS ops are in-order per wave; this waits
  // for outstanding stores before dependent cross-lane loads)
  asm volatile("s_wait_dscnt 0" ::: "memory");
}

// A operand (16x32 bf16): lane holds row m = lane%16; K split 8/8 per half-wave:
//   a[0..7]  = A[m][k0 +      8*half + i]
//   a[8..15] = A[m][k0 + 16 + 8*half + i]
__device__ __forceinline__ bf16x16 load_a(const __bf16* row, int k0, int half) {
  bf16x8 lo = *(const bf16x8*)(row + k0 + 8 * half);
  bf16x8 hi = *(const bf16x8*)(row + k0 + 16 + 8 * half);
  return __builtin_shufflevector(lo, hi, 0,1,2,3,4,5,6,7,8,9,10,11,12,13,14,15);
}

// B operand (32x16 bf16): lane holds col n = lane%16; K split 8/8 per half-wave:
//   b[0..7] = W[k0 + 8*half + i][n], b[8..15] = W[k0 + 16 + 8*half + i][n]   (W stored transposed [n][k] in LDS)
__device__ __forceinline__ bf16x16 load_b(const __bf16* rowT, int k0, int half) {
  bf16x8 lo = *(const bf16x8*)(rowT + k0 + 8 * half);
  bf16x8 hi = *(const bf16x8*)(rowT + k0 + 16 + 8 * half);
  return __builtin_shufflevector(lo, hi, 0,1,2,3,4,5,6,7,8,9,10,11,12,13,14,15);
}

__device__ __forceinline__ bf16x4 pack4(float a, float b, float c, float d) {
  bf16x4 r; r[0] = (__bf16)a; r[1] = (__bf16)b; r[2] = (__bf16)c; r[3] = (__bf16)d;
  return r;
}

// Native no-return float atomic max (GLOBAL_ATOMIC_MAX_NUM_F32, STOREcnt path).
// Device scope: the reduction crosses WGPs, so the RMW must resolve in L2.
__device__ __forceinline__ void atomic_max_f32(float* p, float v) {
  asm volatile("global_atomic_max_num_f32 %0, %1, off scope:SCOPE_DEV"
               :: "v"(p), "v"(v) : "memory");
}

__global__ void __launch_bounds__(256, 1)
edgeconv_kernel(const float* __restrict__ x,      // [N,128] current features
                const int*   __restrict__ esrc,   // [E] source (neighbor j)
                const int*   __restrict__ edst,   // [E] dest   (center  i)
                const float* __restrict__ W1,     // [256,128] row-major
                const float* __restrict__ b1,     // [128]
                const float* __restrict__ W2,     // [128,128] row-major
                const float* __restrict__ b2,     // [128]
                float*       __restrict__ agg,    // [N,128] pre-filled -FLT_MAX
                int nTiles, int nEdges)
{
  extern __shared__ char smem[];
  __bf16* W1t    = (__bf16*)smem;                 // [128][W1_STR] (transposed)
  __bf16* W2t    = W1t + 128 * W1_STR;            // [128][W2_STR] (transposed)
  __bf16* Atiles = W2t + 128 * W2_STR;            // per-wave [16][A_STR]
  __bf16* Htiles = Atiles + WAVES_PB * 16 * A_STR;// per-wave [16][H_STR]

  // ---- stage weights (transpose to [n][k], fp32 -> bf16), once per block ----
  for (int i = threadIdx.x; i < KDIM * CH; i += blockDim.x) {
    int k = i >> 7, n = i & 127;                  // W1[k][n] at k*128+n
    W1t[n * W1_STR + k] = (__bf16)W1[i];
  }
  for (int i = threadIdx.x; i < CH * CH; i += blockDim.x) {
    int k = i >> 7, n = i & 127;
    W2t[n * W2_STR + k] = (__bf16)W2[i];
  }
  __syncthreads();

  const int lane   = threadIdx.x & 31;
  const int waveId = threadIdx.x >> 5;
  const int half   = lane >> 4;                   // 0 / 1 (wave32 half)
  const int nl     = lane & 15;                   // col / row within tile
  const int gwave  = blockIdx.x * WAVES_PB + waveId;
  const int stride = gridDim.x * WAVES_PB;

  __bf16* Atile = Atiles + waveId * 16 * A_STR;
  __bf16* Htile = Htiles + waveId * 16 * H_STR;

  // tile-invariant per-lane biases: hoist out of the hot loop
  float bias1[8], bias2[8];
#pragma unroll
  for (int nt = 0; nt < 8; ++nt) {
    bias1[nt] = b1[nt * 16 + nl];
    bias2[nt] = b2[nt * 16 + nl];
  }

  for (int t = gwave; t < nTiles; t += stride) {
    const int e0 = t * 16;

    // ---- gather: A[m] = [ x_i | x_j - x_i ] as bf16 (2 lanes per edge) ----
    {
      int m  = lane >> 1;
      int ch = (lane & 1) * 64;
      long e = e0 + m; if (e >= nEdges) e = nEdges - 1;
      const float* xi = x + (long)edst[e] * CH;   // center i = dst
      const float* xj = x + (long)esrc[e] * CH;   // neighbor j = src
      __bf16* arow = Atile + m * A_STR;
#pragma unroll
      for (int c = 0; c < 64; c += 4) {
        float4 vi = *(const float4*)(xi + ch + c);
        float4 vj = *(const float4*)(xj + ch + c);
        *(bf16x4*)(arow + ch + c)      = pack4(vi.x, vi.y, vi.z, vi.w);
        *(bf16x4*)(arow + CH + ch + c) = pack4(vj.x - vi.x, vj.y - vi.y,
                                               vj.z - vi.z, vj.w - vi.w);
      }
    }
    lds_fence();

    // dst indices for this lane's 8 output rows (m = r + 8*half)
    int dsts[8];  bool valid[8];
#pragma unroll
    for (int r = 0; r < 8; ++r) {
      int e = e0 + r + 8 * half;
      valid[r] = (e < nEdges);
      dsts[r]  = edst[valid[r] ? e : 0];
    }

    // ---- GEMM1: H[16x128] = relu(A[16x256] @ W1 + b1), bf16 out to LDS ----
    const __bf16* arow = Atile + nl * A_STR;      // A row m = lane%16
#pragma unroll
    for (int nt = 0; nt < 8; ++nt) {
      f32x8 acc = {};
      const __bf16* wrow = W1t + (nt * 16 + nl) * W1_STR;
#pragma unroll
      for (int kc = 0; kc < 8; ++kc) {
        bf16x16 a = load_a(arow, kc * 32, half);
        bf16x16 b = load_b(wrow, kc * 32, half);
        acc = __builtin_amdgcn_wmma_f32_16x16x32_bf16(
                  false, a, false, b, (short)0, acc, false, false);
      }
#pragma unroll
      for (int r = 0; r < 8; ++r) {
        float v = acc[r] + bias1[nt];
        v = v > 0.f ? v : 0.f;                    // ReLU
        Htile[(r + 8 * half) * H_STR + nt * 16 + nl] = (__bf16)v;
      }
    }
    lds_fence();

    // ---- GEMM2: out[16x128] = H @ W2 + b2, scatter atomic-max by dst ----
    const __bf16* hrow = Htile + nl * H_STR;
#pragma unroll
    for (int nt = 0; nt < 8; ++nt) {
      f32x8 acc = {};
      const __bf16* wrow = W2t + (nt * 16 + nl) * W2_STR;
#pragma unroll
      for (int kc = 0; kc < 4; ++kc) {
        bf16x16 a = load_a(hrow, kc * 32, half);
        bf16x16 b = load_b(wrow, kc * 32, half);
        acc = __builtin_amdgcn_wmma_f32_16x16x32_bf16(
                  false, a, false, b, (short)0, acc, false, false);
      }
#pragma unroll
      for (int r = 0; r < 8; ++r) {
        if (valid[r]) {
          float v = acc[r] + bias2[nt];
          atomic_max_f32(agg + (long)dsts[r] * CH + nt * 16 + nl, v);
        }
      }
    }
  }
}

__global__ void fill_kernel(float* __restrict__ p, long n, float v) {
  long i = (long)blockIdx.x * blockDim.x + threadIdx.x;
  if (i < n) p[i] = v;
}

// relu=1: next_x = max(agg, 0)   (folds the "no-edge -> 0" fill AND the ReLU)
// relu=0: out    = (agg == -FLT_MAX) ? 0 : agg   (final layer, no ReLU)
__global__ void finalize_kernel(const float* __restrict__ agg,
                                float* __restrict__ out, long n, int relu) {
  long i = (long)blockIdx.x * blockDim.x + threadIdx.x;
  if (i < n) {
    float v = agg[i];
    out[i] = relu ? (v > 0.f ? v : 0.f) : (v == NEG_HUGE ? 0.f : v);
  }
}

extern "C" void kernel_launch(void* const* d_in, const int* in_sizes, int n_in,
                              void* d_out, int out_size, void* d_ws, size_t ws_size,
                              hipStream_t stream) {
  const float* x    = (const float*)d_in[0];
  const int*   eidx = (const int*)d_in[1];
  const float* W1i = (const float*)d_in[2];  const float* b1i = (const float*)d_in[3];
  const float* W2i = (const float*)d_in[4];  const float* b2i = (const float*)d_in[5];
  const float* W1h = (const float*)d_in[6];  const float* b1h = (const float*)d_in[7];
  const float* W2h = (const float*)d_in[8];  const float* b2h = (const float*)d_in[9];
  const float* W1o = (const float*)d_in[10]; const float* b1o = (const float*)d_in[11];
  const float* W2o = (const float*)d_in[12]; const float* b2o = (const float*)d_in[13];

  const int  N = in_sizes[0] / CH;        // 50000
  const int  E = in_sizes[1] / 2;         // 800000
  const int* esrc = eidx;
  const int* edst = eidx + E;
  const long elems = (long)N * CH;

  // workspace: agg | xa | xb  (each N*128 f32)
  float* agg = (float*)d_ws;
  float* xa  = agg + elems;
  float* xb  = xa + elems;

  const int nTiles  = (E + 15) / 16;                 // 50000
  const int fillBlk = (int)((elems + 255) / 256);
  int waves  = (nTiles + 7) / 8;                     // ~8 tiles per wave
  int eBlk   = (waves + WAVES_PB - 1) / WAVES_PB;

  (void)hipFuncSetAttribute((const void*)edgeconv_kernel,
                            hipFuncAttributeMaxDynamicSharedMemorySize, LDS_BYTES);

  // ---- layer 1 ----
  fill_kernel<<<fillBlk, 256, 0, stream>>>(agg, elems, NEG_HUGE);
  edgeconv_kernel<<<eBlk, 256, LDS_BYTES, stream>>>(x, esrc, edst,
      W1i, b1i, W2i, b2i, agg, nTiles, E);
  finalize_kernel<<<fillBlk, 256, 0, stream>>>(agg, xa, elems, 1);

  // ---- layer 2 ----
  fill_kernel<<<fillBlk, 256, 0, stream>>>(agg, elems, NEG_HUGE);
  edgeconv_kernel<<<eBlk, 256, LDS_BYTES, stream>>>(xa, esrc, edst,
      W1h, b1h, W2h, b2h, agg, nTiles, E);
  finalize_kernel<<<fillBlk, 256, 0, stream>>>(agg, xb, elems, 1);

  // ---- layer 3 ----
  fill_kernel<<<fillBlk, 256, 0, stream>>>(agg, elems, NEG_HUGE);
  edgeconv_kernel<<<eBlk, 256, LDS_BYTES, stream>>>(xb, esrc, edst,
      W1o, b1o, W2o, b2o, agg, nTiles, E);
  finalize_kernel<<<fillBlk, 256, 0, stream>>>(agg, (float*)d_out, elems, 0);
}
